// TimeHypLSTMEncoder_28819230556829
// MI455X (gfx1250) — compile-verified, matrix operand order checked by reading, not run on hardware
//
#include <hip/hip_runtime.h>
#include <cstdint>
#include <cstddef>

// ---------------------------------------------------------------------------
// Optional gfx1250 async global->LDS path (ASYNCcnt). Guarded so the file
// compiles identically to the sync version if the builtins are absent.
// Probe round 3 showed the builtin exists with arg0 = global int4*, so the
// prototype is (global int4*, local int4*, imm offset, imm cpol).
// ---------------------------------------------------------------------------
#if defined(__has_builtin)
# if __has_builtin(__builtin_amdgcn_global_load_async_to_lds_b128) && \
     __has_builtin(__builtin_amdgcn_s_wait_asynccnt)
#  define ATHENA_ASYNC_LDS 1
# endif
#endif

typedef __attribute__((ext_vector_type(4))) int v4i;
#ifdef ATHENA_ASYNC_LDS
typedef __attribute__((address_space(1))) v4i gv4i;   // global int4
typedef __attribute__((address_space(3))) v4i lv4i;   // LDS int4
#endif

// ---------------------------------------------------------------------------
// Problem constants (from reference setup_inputs)
// ---------------------------------------------------------------------------
constexpr int BATCH = 512;
constexpr int SEQ   = 75;
constexpr int DIM   = 768;
constexpr int HID   = 512;
constexpr int HID4  = 4 * HID;   // 2048

constexpr float MIN_NORM = 1e-15f;
constexpr float PROJ_EPS = 4e-3f;

typedef __attribute__((ext_vector_type(16))) __bf16 v16bf;
typedef __attribute__((ext_vector_type(8)))  __bf16 v8bf;
typedef __attribute__((ext_vector_type(8)))  float  v8f;

// ---------------------------------------------------------------------------
// Small math helpers
// ---------------------------------------------------------------------------
static __device__ __forceinline__ float artanh_c(float x) {
    const float CL = 1.0f - 1e-7f;
    x = fminf(fmaxf(x, -CL), CL);
    return 0.5f * logf((1.0f + x) / (1.0f - x));
}

static __device__ __forceinline__ float sigmoidf(float x) {
    return 1.0f / (1.0f + expf(-x));
}

// Block-wide sum reduction of N values; blocks are 256 threads (8 waves).
// sbuf must hold >= 9*N floats (we size it 128 for N up to 13).
template<int N>
static __device__ __forceinline__ void block_reduce(float* v, float* sbuf) {
    const int lane = threadIdx.x & 31;
    const int wid  = threadIdx.x >> 5;
#pragma unroll
    for (int i = 0; i < N; ++i) {
#pragma unroll
        for (int off = 16; off > 0; off >>= 1)
            v[i] += __shfl_down(v[i], off, 32);
    }
    if (lane == 0) {
#pragma unroll
        for (int i = 0; i < N; ++i) sbuf[wid * N + i] = v[i];
    }
    __syncthreads();
    if (threadIdx.x < N) {
        float s = 0.0f;
#pragma unroll
        for (int w = 0; w < 8; ++w) s += sbuf[w * N + threadIdx.x];
        sbuf[8 * N + threadIdx.x] = s;
    }
    __syncthreads();
#pragma unroll
    for (int i = 0; i < N; ++i) v[i] = sbuf[8 * N + i];
    __syncthreads();
}

// mobius_add vector combine given precomputed scalars
static __device__ __forceinline__ void mobius_combine(
    const float x[2], const float y[2],
    float x2, float y2, float xy, float out[2]) {
    float ca  = 1.0f + 2.0f * xy + y2;
    float cb  = 1.0f - x2;
    float den = fmaxf(1.0f + 2.0f * xy + x2 * y2, MIN_NORM);
    out[0] = (ca * x[0] + cb * y[0]) / den;
    out[1] = (ca * x[1] + cb * y[1]) / den;
}

// mobius_pointwise_mul factor from precomputed sums ||x||^2 and ||w.x||^2
static __device__ __forceinline__ float mpm_factor(float x2sum, float wx2sum,
                                                   bool* zer) {
    float xn_t  = sqrtf(x2sum);
    float wxn_t = sqrtf(wx2sum);
    float xn  = fmaxf(xn_t,  MIN_NORM);
    float wxn = fmaxf(wxn_t, MIN_NORM);
    *zer = (wxn_t == 0.0f);
    return tanhf(wxn / xn * artanh_c(xn)) / wxn;
}

// ---------------------------------------------------------------------------
// WMMA bf16 GEMM, LDS double buffer, optionally serving two independent
// problems in one grid (blocks partitioned along x):
//   problem0: C0[M,N0] = A0[M,K] * B0[K,N0] + bias0   (x-tiles [0, N0/BN) )
//   problem1: C1[M,N1] = A1[M,K] * B1[K,N1] + bias1   (x-tiles [N0/BN, ...))
// A tile: async global->LDS when available (frees staging VGPRs), else
// register-staged. B tile: register-staged (transposed on LDS store so both
// fragments are contiguous ds_load_b128 per the CDNA5 16-bit layout).
// ---------------------------------------------------------------------------
template<int BM, int BN, int BK, int WR, int WC>
__global__ __launch_bounds__(32 * WR * WC)
void gemm_bf16_dual(const __bf16* __restrict__ A0, const __bf16* __restrict__ B0,
                    const float* __restrict__ bias0, float* __restrict__ C0, int N0,
                    const __bf16* __restrict__ A1, const __bf16* __restrict__ B1,
                    const float* __restrict__ bias1, float* __restrict__ C1, int N1,
                    int M, int K)
{
    constexpr int NT  = 32 * WR * WC;
    constexpr int PAD = 8;                 // rows 16B aligned (BK+8=72 -> 144B)
    constexpr int LDT = BK + PAD;
    constexpr int WTM = BM / WR, WTN = BN / WC;
    constexpr int TM  = WTM / 16, TN = WTN / 16;
    constexpr int APT = (BM * BK / 2) / NT;   // u32 per thread for A slab
    constexpr int BPT = (BK * BN / 2) / NT;   // u32 per thread for B slab
    constexpr int ACH = (BM * BK / 8) / NT;   // 16B chunks per thread for A

    __shared__ __bf16 As[2][BM][LDT];
    __shared__ __bf16 Bs[2][BN][LDT];      // transposed: Bs[buf][n][k]

    // ---- per-block problem selection (uniform) ----
    const int xt0 = N0 / BN;
    const __bf16* A; const __bf16* Bm; const float* bias; float* C; int N; int bx;
    if ((int)blockIdx.x < xt0) {
        A = A0; Bm = B0; bias = bias0; C = C0; N = N0; bx = blockIdx.x;
    } else {
        A = A1; Bm = B1; bias = bias1; C = C1; N = N1; bx = blockIdx.x - xt0;
    }

    const int tid  = threadIdx.x;
    const int lane = tid & 31;
    const int wid  = tid >> 5;
    const int wr   = wid % WR;
    const int wc   = wid / WR;
    const int m0   = blockIdx.y * BM;
    const int n0   = bx * BN;

    const v8f vzero = {0.f, 0.f, 0.f, 0.f, 0.f, 0.f, 0.f, 0.f};
    v8f acc[TM][TN];
#pragma unroll
    for (int ti = 0; ti < TM; ++ti)
#pragma unroll
        for (int tj = 0; tj < TN; ++tj) acc[ti][tj] = vzero;

    union U32 { uint32_t u; __bf16 h[2]; };
    union Frag { v16bf v; v8bf h[2]; };

#ifndef ATHENA_ASYNC_LDS
    uint32_t ra[APT];
#endif
    uint32_t rb[BPT];

#ifdef ATHENA_ASYNC_LDS
    // async DMA of the A tile straight into LDS (tracked by ASYNCcnt)
    auto async_load_A = [&](int buf, int k0) {
#pragma unroll
        for (int j = 0; j < ACH; ++j) {
            int idx = tid + j * NT;
            int r  = idx / (BK / 8);
            int c8 = (idx % (BK / 8)) * 8;
            const __bf16* gp = A + (size_t)(m0 + r) * K + k0 + c8;
            __bf16* lp = &As[buf][r][c8];
            __builtin_amdgcn_global_load_async_to_lds_b128(
                (gv4i*)(uintptr_t)gp,
                (lv4i*)(uint32_t)(uintptr_t)lp,
                0, 0);
        }
    };
#else
    auto load_regs_A = [&](int k0) {
#pragma unroll
        for (int j = 0; j < APT; ++j) {
            int idx = tid + j * NT;
            int r = idx / (BK / 2);
            int c = idx % (BK / 2);
            ra[j] = *reinterpret_cast<const uint32_t*>(
                A + (size_t)(m0 + r) * K + k0 + 2 * c);
        }
    };
    auto store_A = [&](int buf) {
#pragma unroll
        for (int j = 0; j < APT; ++j) {
            int idx = tid + j * NT;
            int r = idx / (BK / 2);
            int c = idx % (BK / 2);
            *reinterpret_cast<uint32_t*>(&As[buf][r][2 * c]) = ra[j];
        }
    };
#endif

    auto load_regs_B = [&](int k0) {
#pragma unroll
        for (int j = 0; j < BPT; ++j) {
            int idx = tid + j * NT;
            int r = idx / (BN / 2);          // k
            int c = idx % (BN / 2);          // n/2
            rb[j] = *reinterpret_cast<const uint32_t*>(
                Bm + (size_t)(k0 + r) * N + n0 + 2 * c);
        }
        // speculative prefetch of the slab after this one
        int kpf = k0 + BK;
        if (kpf < K) {
            __builtin_prefetch(A + (size_t)(m0 + (tid % BM)) * K + kpf, 0, 1);
            __builtin_prefetch(Bm + (size_t)(kpf + (tid % BK)) * N + n0, 0, 1);
        }
    };

    auto store_B = [&](int buf) {
#pragma unroll
        for (int j = 0; j < BPT; ++j) {
            int idx = tid + j * NT;
            int r = idx / (BN / 2);
            int c = idx % (BN / 2);
            U32 u; u.u = rb[j];
            Bs[buf][2 * c    ][r] = u.h[0];
            Bs[buf][2 * c + 1][r] = u.h[1];
        }
    };

    const int nslab = K / BK;
#ifdef ATHENA_ASYNC_LDS
    async_load_A(0, 0);
#else
    load_regs_A(0);
#endif
    load_regs_B(0);
#ifndef ATHENA_ASYNC_LDS
    store_A(0);
#endif
    store_B(0);

    const int half = lane >> 4;
    const int l15  = lane & 15;

    for (int s = 0; s < nslab; ++s) {
        const int cur = s & 1;
        if (s + 1 < nslab) load_regs_B((s + 1) * BK);  // global latency overlaps
#ifdef ATHENA_ASYNC_LDS
        __builtin_amdgcn_s_wait_asynccnt(0);           // our async A writes done
#endif
        __syncthreads();                               // buf 'cur' ready
#pragma unroll
        for (int kk = 0; kk < BK; kk += 32) {
            Frag af[TM], bfr[TN];
#pragma unroll
            for (int ti = 0; ti < TM; ++ti) {
                int row = wr * WTM + ti * 16 + l15;
                af[ti].h[0] = *reinterpret_cast<const v8bf*>(&As[cur][row][kk + half * 8]);
                af[ti].h[1] = *reinterpret_cast<const v8bf*>(&As[cur][row][kk + 16 + half * 8]);
            }
#pragma unroll
            for (int tj = 0; tj < TN; ++tj) {
                int col = wc * WTN + tj * 16 + l15;
                bfr[tj].h[0] = *reinterpret_cast<const v8bf*>(&Bs[cur][col][kk + half * 8]);
                bfr[tj].h[1] = *reinterpret_cast<const v8bf*>(&Bs[cur][col][kk + 16 + half * 8]);
            }
#pragma unroll
            for (int ti = 0; ti < TM; ++ti)
#pragma unroll
                for (int tj = 0; tj < TN; ++tj)
                    acc[ti][tj] = __builtin_amdgcn_wmma_f32_16x16x32_bf16(
                        false, af[ti].v, false, bfr[tj].v,
                        (short)0, acc[ti][tj], false, false);
        }
        if (s + 1 < nslab) {
            __syncthreads();                           // all done reading 'cur^1'
#ifndef ATHENA_ASYNC_LDS
            store_A(cur ^ 1);
#endif
            store_B(cur ^ 1);
#ifdef ATHENA_ASYNC_LDS
            async_load_A(cur ^ 1, (s + 1) * BK);
#endif
        }
    }

    // Epilogue: lane l -> col = l&15, rows = r + 8*(l>>4); add bias
#pragma unroll
    for (int tj = 0; tj < TN; ++tj) {
        int n = n0 + wc * WTN + tj * 16 + l15;
        float bv = bias[n];
#pragma unroll
        for (int ti = 0; ti < TM; ++ti) {
            int mb = m0 + wr * WTM + ti * 16 + half * 8;
#pragma unroll
            for (int r = 0; r < 8; ++r)
                C[(size_t)(mb + r) * N + n] = acc[ti][tj][r] + bv;
        }
    }
}

// ---------------------------------------------------------------------------
// Encoder: x = project(expmap0(feats)), write bf16 for the GEMM.
// One 256-thread block per row of D=768.
// ---------------------------------------------------------------------------
__global__ __launch_bounds__(256)
void encode_kernel(const float* __restrict__ x, __bf16* __restrict__ out)
{
    __shared__ float sbuf[128];
    const size_t row = blockIdx.x;
    const int tid = threadIdx.x;
    const float* p = x + row * DIM;
    float v[3];
    float s = 0.0f;
#pragma unroll
    for (int j = 0; j < 3; ++j) { v[j] = p[tid + j * 256]; s += v[j] * v[j]; }
    float red[1] = { s };
    block_reduce<1>(red, sbuf);
    float nt = sqrtf(red[0]);
    float n  = fmaxf(nt, MIN_NORM);
    float f  = tanhf(n) / n;                 // expmap0 scale
    float ny = fmaxf(f * nt, MIN_NORM);      // norm after expmap0
    float maxn = 1.0f - PROJ_EPS;
    if (ny > maxn) f *= maxn / ny;           // project
    __bf16* q = out + row * DIM;
#pragma unroll
    for (int j = 0; j < 3; ++j) q[tid + j * 256] = (__bf16)(f * v[j]);
}

__global__ void convert_bf16_kernel(const float* __restrict__ src,
                                    __bf16* __restrict__ dst, int n)
{
    for (int i = blockIdx.x * blockDim.x + threadIdx.x; i < n;
         i += gridDim.x * blockDim.x)
        dst[i] = (__bf16)src[i];
}

__global__ void init_state_kernel(float* hf, float* cf,
                                  __bf16* hb, __bf16* cb, int n)
{
    for (int i = blockIdx.x * blockDim.x + threadIdx.x; i < n;
         i += gridDim.x * blockDim.x) {
        hf[i] = 0.0f; cf[i] = 0.0f;
        hb[i] = (__bf16)0.0f; cb[i] = (__bf16)0.0f;
    }
}

// ---------------------------------------------------------------------------
// One recurrence step. 256 threads per batch row, 2 elements each (H=512).
// Reductions fused into 9 block_reduce calls (max width 13).
// ---------------------------------------------------------------------------
__global__ __launch_bounds__(256)
void step_kernel(const float* __restrict__ tmp_d,   // cc @ W_d + b   [B,H]
                 const float* __restrict__ w_t,     // h @ W_all + b  [B,4H]
                 const float* __restrict__ useq,    // U_seq          [B,S,4H]
                 const float* __restrict__ time_f,  //                [B,S]
                 int t,
                 float* __restrict__ h_f, float* __restrict__ c_f,
                 __bf16* __restrict__ h_b, __bf16* __restrict__ c_b)
{
    __shared__ float sbuf[128];
    const int b   = blockIdx.x;
    const int tid = threadIdx.x;
    const int e0 = tid, e1 = tid + 256;
    const size_t rH = (size_t)b * HID;
    const size_t r4 = (size_t)b * HID4;
    const size_t rU = ((size_t)b * SEQ + t) * HID4;
    const float  tt = time_f[(size_t)b * SEQ + t];

    float d[2]  = { tmp_d[rH + e0], tmp_d[rH + e1] };
    float cc[2] = { c_f[rH + e0],  c_f[rH + e1]  };
    float wv[4][2], uv[4][2];
#pragma unroll
    for (int gi = 0; gi < 4; ++gi) {
        wv[gi][0] = w_t[r4 + gi * HID + e0];
        wv[gi][1] = w_t[r4 + gi * HID + e1];
        uv[gi][0] = useq[rU + gi * HID + e0];
        uv[gi][1] = useq[rU + gi * HID + e1];
    }

    // ---- reduce #1: ||d||^2 + all 4 gates' {||W||^2, ||U||^2, <W,U>} ----
    float r13[13];
    r13[0] = d[0] * d[0] + d[1] * d[1];
#pragma unroll
    for (int gi = 0; gi < 4; ++gi) {
        r13[1 + 3 * gi] = wv[gi][0] * wv[gi][0] + wv[gi][1] * wv[gi][1];
        r13[2 + 3 * gi] = uv[gi][0] * uv[gi][0] + uv[gi][1] * uv[gi][1];
        r13[3 + 3 * gi] = wv[gi][0] * uv[gi][0] + wv[gi][1] * uv[gi][1];
    }
    block_reduce<13>(r13, sbuf);

    float nd = fmaxf(sqrtf(r13[0]), MIN_NORM);
    float fl = artanh_c(nd) / nd;
    float th[2] = { tanhf(fl * d[0]), tanhf(fl * d[1]) };

    float ma[4][2];
#pragma unroll
    for (int gi = 0; gi < 4; ++gi)
        mobius_combine(wv[gi], uv[gi],
                       r13[1 + 3 * gi], r13[2 + 3 * gi], r13[3 + 3 * gi],
                       ma[gi]);

    // ---- reduce #2: ||tanh||^2 + 4 gate mobius-sum norms ----
    float r5[5];
    r5[0] = th[0] * th[0] + th[1] * th[1];
#pragma unroll
    for (int gi = 0; gi < 4; ++gi)
        r5[1 + gi] = ma[gi][0] * ma[gi][0] + ma[gi][1] * ma[gi][1];
    block_reduce<5>(r5, sbuf);

    float nth_t = sqrtf(r5[0]);
    float nth   = fmaxf(nth_t, MIN_NORM);
    float fe    = tanhf(nth) / nth;
    float cs1[2] = { fe * th[0], fe * th[1] };
    float n_cs1  = fe * nth_t;               // true ||c_s1||

    float g[4][2];                           // f, i, o, ct
#pragma unroll
    for (int gi = 0; gi < 4; ++gi) {
        float nm = fmaxf(sqrtf(r5[1 + gi]), MIN_NORM);
        float fg = artanh_c(nm) / nm;
        g[gi][0] = sigmoidf(fg * ma[gi][0]);
        g[gi][1] = sigmoidf(fg * ma[gi][1]);
    }

    // ---- c_s2 = mobius_pointwise_mul(c_s1, t)  (scalar weight) ----
    float xn    = fmaxf(n_cs1, MIN_NORM);
    float wxn_t = fabsf(tt) * n_cs1;
    float wxn   = fmaxf(wxn_t, MIN_NORM);
    float fac   = tanhf(wxn / xn * artanh_c(xn)) / wxn;
    bool  zer   = (wxn_t == 0.0f);
    float cs2[2] = { zer ? 0.0f : fac * tt * cs1[0],
                     zer ? 0.0f : fac * tt * cs1[1] };

    // ---- reduce #3: c_l inputs {||cc||^2, <cs1,cc>} + p1 inputs {||ct||^2, ||i.ct||^2}
    float ict[2] = { g[1][0] * g[3][0], g[1][1] * g[3][1] };
    float r4v[4];
    r4v[0] = cc[0] * cc[0] + cc[1] * cc[1];
    r4v[1] = cs1[0] * cc[0] + cs1[1] * cc[1];
    r4v[2] = g[3][0] * g[3][0] + g[3][1] * g[3][1];
    r4v[3] = ict[0] * ict[0] + ict[1] * ict[1];
    block_reduce<4>(r4v, sbuf);

    float neg_cs1[2] = { -cs1[0], -cs1[1] };
    float cl[2];
    mobius_combine(neg_cs1, cc, n_cs1 * n_cs1, r4v[0], -r4v[1], cl);

    bool z1;
    float f1 = mpm_factor(r4v[2], r4v[3], &z1);
    float p1[2] = { z1 ? 0.0f : f1 * ict[0], z1 ? 0.0f : f1 * ict[1] };

    // ---- reduce #4: c_adj = mobius_add(cl, cs2) ----
    float r3[3];
    r3[0] = cl[0] * cl[0] + cl[1] * cl[1];
    r3[1] = cs2[0] * cs2[0] + cs2[1] * cs2[1];
    r3[2] = cl[0] * cs2[0] + cl[1] * cs2[1];
    block_reduce<3>(r3, sbuf);
    float cadj[2];
    mobius_combine(cl, cs2, r3[0], r3[1], r3[2], cadj);

    // ---- reduce #5: p2 = mpm(f, c_adj) ----
    float fca[2] = { g[0][0] * cadj[0], g[0][1] * cadj[1] };
    float r2[2];
    r2[0] = cadj[0] * cadj[0] + cadj[1] * cadj[1];
    r2[1] = fca[0] * fca[0] + fca[1] * fca[1];
    block_reduce<2>(r2, sbuf);
    bool z2;
    float f2 = mpm_factor(r2[0], r2[1], &z2);
    float p2[2] = { z2 ? 0.0f : f2 * fca[0], z2 ? 0.0f : f2 * fca[1] };

    // ---- reduce #6: cc_new = mobius_add(p1, p2) ----
    r3[0] = p1[0] * p1[0] + p1[1] * p1[1];
    r3[1] = p2[0] * p2[0] + p2[1] * p2[1];
    r3[2] = p1[0] * p2[0] + p1[1] * p2[1];
    block_reduce<3>(r3, sbuf);
    float ccn[2];
    mobius_combine(p1, p2, r3[0], r3[1], r3[2], ccn);

    // ---- reduce #7: logmap0(cc_new) ----
    float r1[1] = { ccn[0] * ccn[0] + ccn[1] * ccn[1] };
    block_reduce<1>(r1, sbuf);
    float nc  = fmaxf(sqrtf(r1[0]), MIN_NORM);
    float fl2 = artanh_c(nc) / nc;
    float q[2] = { tanhf(fl2 * ccn[0]), tanhf(fl2 * ccn[1]) };

    // ---- reduce #8: expmap0(tanh(...)) ----
    r1[0] = q[0] * q[0] + q[1] * q[1];
    block_reduce<1>(r1, sbuf);
    float nq  = fmaxf(sqrtf(r1[0]), MIN_NORM);
    float fe2 = tanhf(nq) / nq;
    float eq[2] = { fe2 * q[0], fe2 * q[1] };

    // ---- reduce #9: h_new = mpm(o, eq) ----
    float oeq[2] = { g[2][0] * eq[0], g[2][1] * eq[1] };
    r2[0] = eq[0] * eq[0] + eq[1] * eq[1];
    r2[1] = oeq[0] * oeq[0] + oeq[1] * oeq[1];
    block_reduce<2>(r2, sbuf);
    bool z3;
    float f3 = mpm_factor(r2[0], r2[1], &z3);
    float hn[2] = { z3 ? 0.0f : f3 * oeq[0], z3 ? 0.0f : f3 * oeq[1] };

    // ---- write state (fp32 for math, bf16 for next step's WMMA GEMMs) ----
    c_f[rH + e0] = ccn[0]; c_f[rH + e1] = ccn[1];
    h_f[rH + e0] = hn[0];  h_f[rH + e1] = hn[1];
    c_b[rH + e0] = (__bf16)ccn[0]; c_b[rH + e1] = (__bf16)ccn[1];
    h_b[rH + e0] = (__bf16)hn[0];  h_b[rH + e1] = (__bf16)hn[1];
}

// ---------------------------------------------------------------------------
// Final: out = concat(logmap0(h), logmap0(c)); one block per output row.
// ---------------------------------------------------------------------------
__global__ __launch_bounds__(256)
void final_logmap_kernel(const float* __restrict__ h_f,
                         const float* __restrict__ c_f,
                         float* __restrict__ out)
{
    __shared__ float sbuf[128];
    const int bi  = blockIdx.x;
    const int tid = threadIdx.x;
    const float* src;
    float* dst;
    if (bi < BATCH) {
        src = h_f + (size_t)bi * HID;
        dst = out + (size_t)bi * HID;
    } else {
        int b2 = bi - BATCH;
        src = c_f + (size_t)b2 * HID;
        dst = out + (size_t)BATCH * HID + (size_t)b2 * HID;
    }
    float v[2] = { src[tid], src[tid + 256] };
    float red[1] = { v[0] * v[0] + v[1] * v[1] };
    block_reduce<1>(red, sbuf);
    float n = fmaxf(sqrtf(red[0]), MIN_NORM);
    float f = artanh_c(n) / n;
    dst[tid]       = f * v[0];
    dst[tid + 256] = f * v[1];
}

// ---------------------------------------------------------------------------
// Host launch
// ---------------------------------------------------------------------------
extern "C" void kernel_launch(void* const* d_in, const int* in_sizes, int n_in,
                              void* d_out, int out_size, void* d_ws, size_t ws_size,
                              hipStream_t stream)
{
    const float* sent    = (const float*)d_in[0];
    const float* time_f  = (const float*)d_in[1];
    const float* W_all_w = (const float*)d_in[2];
    const float* W_all_b = (const float*)d_in[3];
    const float* U_all_w = (const float*)d_in[4];
    const float* U_all_b = (const float*)d_in[5];
    const float* W_d_w   = (const float*)d_in[6];
    const float* W_d_b   = (const float*)d_in[7];
    float* out = (float*)d_out;

    char* ws = (char*)d_ws;
    size_t off = 0;
    auto take = [&](size_t bytes) -> char* {
        char* p = ws + off;
        off = (off + bytes + 255) & ~(size_t)255;
        return p;
    };

    float*  Useq = (float*) take((size_t)BATCH * SEQ * HID4 * 4);
    __bf16* xb   = (__bf16*)take((size_t)BATCH * SEQ * DIM * 2);
    __bf16* Uwb  = (__bf16*)take((size_t)DIM * HID4 * 2);
    __bf16* Wwb  = (__bf16*)take((size_t)HID * HID4 * 2);
    __bf16* Wdb  = (__bf16*)take((size_t)HID * HID * 2);
    float*  h_f  = (float*) take((size_t)BATCH * HID * 4);
    float*  c_f  = (float*) take((size_t)BATCH * HID * 4);
    __bf16* h_b  = (__bf16*)take((size_t)BATCH * HID * 2);
    __bf16* c_b  = (__bf16*)take((size_t)BATCH * HID * 2);
    float*  tmpd = (float*) take((size_t)BATCH * HID * 4);
    float*  wt   = (float*) take((size_t)BATCH * HID4 * 4);

    // Weight bf16 conversion
    convert_bf16_kernel<<<1024, 256, 0, stream>>>(U_all_w, Uwb, DIM * HID4);
    convert_bf16_kernel<<<1024, 256, 0, stream>>>(W_all_w, Wwb, HID * HID4);
    convert_bf16_kernel<<<512, 256, 0, stream>>>(W_d_w, Wdb, HID * HID);

    // Encoder onto Poincare ball (bf16 for GEMM A)
    encode_kernel<<<BATCH * SEQ, 256, 0, stream>>>(sent, xb);

    // Big input-side GEMM: U_seq = x @ U_all_w + b   [38400 x 2048]
    gemm_bf16_dual<128, 128, 64, 4, 2>
        <<<dim3(HID4 / 128, (BATCH * SEQ) / 128), 256, 0, stream>>>(
            xb, Uwb, U_all_b, Useq, HID4,
            xb, Uwb, U_all_b, Useq, 0,          // unused second problem
            BATCH * SEQ, DIM);

    init_state_kernel<<<512, 256, 0, stream>>>(h_f, c_f, h_b, c_b, BATCH * HID);

    for (int t = 0; t < SEQ; ++t) {
        // fused per-step GEMMs:
        //   tmp_d = cc @ W_d + b   [512 x 512]   (x-tiles 0..7)
        //   w_t   = h  @ W_all + b [512 x 2048]  (x-tiles 8..39)
        gemm_bf16_dual<64, 64, 64, 2, 2>
            <<<dim3(HID / 64 + HID4 / 64, BATCH / 64), 128, 0, stream>>>(
                c_b, Wdb, W_d_b, tmpd, HID,
                h_b, Wwb, W_all_b, wt, HID4,
                BATCH, HID);
        // hyperbolic elementwise update
        step_kernel<<<BATCH, 256, 0, stream>>>(
            tmpd, wt, Useq, time_f, t, h_f, c_f, h_b, c_b);
    }

    final_logmap_kernel<<<2 * BATCH, 256, 0, stream>>>(h_f, c_f, out);
}
